// WeightGenerator_48773648614385
// MI455X (gfx1250) — matrix operand order repaired and checked by reading, not compiled, not run on hardware
//
#include <hip/hip_runtime.h>
#include <hip/hip_bf16.h>

typedef __attribute__((ext_vector_type(16))) __bf16 v16bf;
typedef __attribute__((ext_vector_type(8)))  __bf16 v8bf;
typedef __attribute__((ext_vector_type(8)))  float  v8f;
typedef unsigned int __attribute__((ext_vector_type(4))) uint4v;
typedef int  __attribute__((ext_vector_type(8))) int8v;
typedef int  __attribute__((ext_vector_type(4))) int4v;

#define CCH   128      // input channels
#define UCH   128      // conv1 output channels
#define BGH   256
#define BGW   256
#define INH   64
#define INW   64
#define NINST 8
#define BIGV  100000.0f
// packed weight fragments: 9 taps * 4 c-blocks * 8 n-tiles * 32 lanes * 16 elems
#define PACKW (9*4*8*32*16)
#define TAPW  (4*8*32*16)          // bf16 elems per tap per (hi|lo) = 16384 (32 KB)

// ---------------------------------------------------------------------------
// K1: NCHW f32 -> HWC bf16 hi/lo pair (hi = bf16(x), lo = bf16(x - hi))
// ---------------------------------------------------------------------------
__global__ void cvt_hwc_bf16(const float* __restrict__ src,
                             __bf16* __restrict__ hi, __bf16* __restrict__ lo,
                             int nb, int hh, int ww) {
  long total = (long)nb * CCH * hh * ww;
  long idx = (long)blockIdx.x * blockDim.x + threadIdx.x;
  if (idx >= total) return;
  long pix = idx >> 7;              // /128
  int  ch  = (int)(idx & 127);
  long plane = (long)hh * ww;
  long b = pix / plane;
  long p = pix - b * plane;
  float v = src[(b * CCH + ch) * plane + p];
  __bf16 h = (__bf16)v;
  hi[idx] = h;
  lo[idx] = (__bf16)(v - (float)h);
}

// ---------------------------------------------------------------------------
// K2: pack conv1 weights (U,C,3,3) f32 into WMMA B-fragment order, bf16 hi/lo.
// idx = (((t*4 + kb)*8 + nt)*32 + lane)*16 + e
//   n = nt*16 + lane%16 ; k(channel) = kb*32 + 16*(lane/16) + e ; tap = t
// ---------------------------------------------------------------------------
__global__ void pack_w(const float* __restrict__ w,
                       __bf16* __restrict__ hi, __bf16* __restrict__ lo) {
  int idx = blockIdx.x * blockDim.x + threadIdx.x;
  if (idx >= PACKW) return;
  int e    =  idx        & 15;
  int lane = (idx >> 4)  & 31;
  int nt   = (idx >> 9)  & 7;
  int kb   = (idx >> 12) & 3;
  int t    =  idx >> 14;                 // 0..8
  int n = nt * 16 + (lane & 15);
  int k = kb * 32 + 16 * (lane >> 4) + e;
  float v = w[(n * CCH + k) * 9 + t];
  __bf16 h = (__bf16)v;
  hi[idx] = h;
  lo[idx] = (__bf16)(v - (float)h);
}

// ---------------------------------------------------------------------------
// TDM: DMA one 32 KB contiguous block (4096 x 8B units) global -> LDS.
// D# per CDNA5 ISA: group0 {count=1, lds_addr, global_addr,
// type=2}; group1 {data_size=3(8B), tensor_dim0=tile_dim0=4096, dims1=1,
// stride0=4096}; groups 2/3 zero (<=2D). Tracked by TENSORcnt.
// This toolchain's builtin takes 6 args: (g0, g1, g2, g3, g4, cpol).
// ---------------------------------------------------------------------------
__device__ __forceinline__ void tdm_load_32k(const __bf16* gsrc, unsigned lds_off) {
  unsigned long long ga = (unsigned long long)gsrc;
  uint4v g0;
  g0.x = 1u;                                          // count=1 (valid descriptor)
  g0.y = lds_off;                                     // LDS byte address
  g0.z = (unsigned)(ga & 0xFFFFFFFFu);                // global_addr[31:0]
  g0.w = (unsigned)((ga >> 32) & 0x01FFFFFFu)         // global_addr[56:32]
       | 0x80000000u;                                 // type=2 ("image")
  int8v g1;
  g1[0] = 0x30000;                                    // data_size=3 (8 bytes)
  g1[1] = (int)(4096u << 16);                         // tensor_dim0[15:0]=4096
  g1[2] = (int)(1u << 16);                            // tensor_dim0[31:16]=0, tensor_dim1[15:0]=1
  g1[3] = (int)(4096u << 16);                         // tensor_dim1[31:16]=0, tile_dim0=4096
  g1[4] = 1;                                          // tile_dim1=1, tile_dim2=0
  g1[5] = 4096;                                       // tensor_dim0_stride[31:0]
  g1[6] = 0;                                          // stride0[47:32]=0, stride1 lo=0
  g1[7] = 0;
  int4v z4 = {0, 0, 0, 0};
  int8v z8 = {0, 0, 0, 0, 0, 0, 0, 0};
  __builtin_amdgcn_tensor_load_to_lds(g0, g1, z4, z4, z8, 0);
}

// ---------------------------------------------------------------------------
// K3: conv1 (3x3, SAME, 128->128) + bias + ReLU via v_wmma_f32_16x16x32_bf16.
// Implicit GEMM: M=16 pixels/wave, N=128 (8 tiles), K=1152 (9 taps x 4 c-blks).
// fp32 accuracy recovered with hi/lo bf16 split (3 WMMAs per tile).
// B fragments are TDM-DMA'd per tap into double-buffered LDS (wave 0 issues,
// TENSORcnt + workgroup barrier synchronize), read back with ds_load_b128.
// block = (32,4): 4 waves, each a 16-px strip. grid = (W/64, H, nb).
// ---------------------------------------------------------------------------
__global__ __launch_bounds__(128)
void conv1_wmma(const __bf16* __restrict__ ahi, const __bf16* __restrict__ alo,
                const __bf16* __restrict__ bhi, const __bf16* __restrict__ blo,
                const float* __restrict__ bias, float* __restrict__ outHWC,
                int Hd, int Wd) {
  // [buffer][hi|lo][16384 bf16]  -> 128 KB static LDS (<= 320 KB/WGP)
  __shared__ __attribute__((aligned(32))) __bf16 ldsB[2][2][TAPW];

  const int lane = threadIdx.x;          // 0..31
  const int wv   = threadIdx.y;          // 0..3
  const int y    = blockIdx.y;
  const int b    = blockIdx.z;
  const int x0   = (blockIdx.x * 4 + wv) * 16;
  const int m    = lane & 15;
  const int h    = lane >> 4;
  const int px   = x0 + m;

  v8f acc[8];
  for (int nt = 0; nt < 8; ++nt)
    for (int r = 0; r < 8; ++r) acc[nt][r] = 0.0f;

  // prologue: DMA tap 0's B fragments (hi+lo) into buffer 0
  if (wv == 0) {
    tdm_load_32k(bhi, (unsigned)(unsigned long long)&ldsB[0][0][0]);
    tdm_load_32k(blo, (unsigned)(unsigned long long)&ldsB[0][1][0]);
  }

  for (int t = 0; t < 9; ++t) {
    // all waves done reading the buffer about to be overwritten
    __syncthreads();
    if (wv == 0) {
      if (t < 8) {
        const int nb = (t + 1) & 1;
        tdm_load_32k(bhi + (long)(t + 1) * TAPW,
                     (unsigned)(unsigned long long)&ldsB[nb][0][0]);
        tdm_load_32k(blo + (long)(t + 1) * TAPW,
                     (unsigned)(unsigned long long)&ldsB[nb][1][0]);
        __builtin_amdgcn_s_wait_tensorcnt(2);   // tap t's 2 DMAs complete
      } else {
        __builtin_amdgcn_s_wait_tensorcnt(0);
      }
    }
    __syncthreads();                            // tap t data visible to all waves

    const __bf16* lbh = &ldsB[t & 1][0][0];
    const __bf16* lbl = &ldsB[t & 1][1][0];

    const int dy = t / 3 - 1, dx = t % 3 - 1;
    const int yy = y + dy, xx = px + dx;
    const bool inb = (yy >= 0 && yy < Hd && xx >= 0 && xx < Wd);
    const long pbase = (((long)(b * Hd + yy) * Wd) + xx) << 7;   // *128 ch
    if (t < 8 && inb) {
      // speculative prefetch of next tap's activation row (global_prefetch_b8)
      __builtin_prefetch(ahi + pbase + (long)Wd * CCH, 0, 0);
    }
    for (int kb = 0; kb < 4; ++kb) {
      const int c0 = kb * 32 + 16 * h;
      v16bf Ah, Al;
      if (inb) {
        v8bf h0 = *(const v8bf*)(ahi + pbase + c0);
        v8bf h1 = *(const v8bf*)(ahi + pbase + c0 + 8);
        v8bf l0 = *(const v8bf*)(alo + pbase + c0);
        v8bf l1 = *(const v8bf*)(alo + pbase + c0 + 8);
        for (int e = 0; e < 8; ++e) {
          Ah[e] = h0[e]; Ah[e + 8] = h1[e];
          Al[e] = l0[e]; Al[e + 8] = l1[e];
        }
      } else {
        for (int e = 0; e < 16; ++e) { Ah[e] = (__bf16)0.0f; Al[e] = (__bf16)0.0f; }
      }
      const int wofs = (kb * 8) * 512 + lane * 16;      // per nt add 512
      for (int nt = 0; nt < 8; ++nt) {
        v16bf Bh = *(const v16bf*)(lbh + wofs + nt * 512);   // ds_load_b128 x2
        v16bf Bl = *(const v16bf*)(lbl + wofs + nt * 512);
        acc[nt] = __builtin_amdgcn_wmma_f32_16x16x32_bf16(
            false, Ah, false, Bh, (short)0, acc[nt], false, false);
        acc[nt] = __builtin_amdgcn_wmma_f32_16x16x32_bf16(
            false, Ah, false, Bl, (short)0, acc[nt], false, false);
        acc[nt] = __builtin_amdgcn_wmma_f32_16x16x32_bf16(
            false, Al, false, Bh, (short)0, acc[nt], false, false);
      }
    }
  }

  // D layout: lane -> channel col (lane%16), rows r+8*(lane/16) -> pixels
  for (int nt = 0; nt < 8; ++nt) {
    const int ch = nt * 16 + m;
    const float bv = bias[ch];
    for (int r = 0; r < 8; ++r) {
      float v = acc[nt][r] + bv;
      v = v > 0.0f ? v : 0.0f;
      const int pxo = x0 + r + 8 * h;
      outHWC[((((long)(b * Hd + y) * Wd) + pxo) << 7) + ch] = v;
    }
  }
}

// ---------------------------------------------------------------------------
// K4: conv2 (3x3, SAME, 128->1) + ReLU. VALU dot (tiny: N=1). grid.z = batch.
// ---------------------------------------------------------------------------
__global__ void conv2_relu(const float* __restrict__ inHWC,
                           const float* __restrict__ w2,
                           const float* __restrict__ b2,
                           float* __restrict__ pred, int Hd, int Wd) {
  const int b = blockIdx.z;
  long tid = (long)blockIdx.x * blockDim.x + threadIdx.x;
  if (tid >= (long)Hd * Wd) return;
  const int y = (int)(tid / Wd), x = (int)(tid % Wd);
  float s = b2[0];
  for (int t = 0; t < 9; ++t) {
    const int yy = y + t / 3 - 1, xx = x + t % 3 - 1;
    if (yy < 0 || yy >= Hd || xx < 0 || xx >= Wd) continue;
    const float* p = inHWC + ((((long)(b * Hd + yy) * Wd) + xx) << 7);
    for (int c = 0; c < CCH; ++c) s += p[c] * w2[c * 9 + t];
  }
  s = s > 0.0f ? s : 0.0f;
  pred[(long)b * Hd * Wd + tid] = s;
}

// ---------------------------------------------------------------------------
// K5: fused resize + pad + softmax(9) + weighted feature blend.
// One thread per (pixel, 8-channel group). jax.image.resize bilinear
// (half-pixel centers, edge-clamped taps).
// ---------------------------------------------------------------------------
__global__ void fuse_out(const float* __restrict__ instf,
                         const float* __restrict__ bgf,
                         const int* __restrict__ bbox,
                         const float* __restrict__ predI,
                         const float* __restrict__ predBG,
                         float* __restrict__ out) {
  long tid = (long)blockIdx.x * blockDim.x + threadIdx.x;
  if (tid >= (long)BGH * BGW * 16) return;
  const int cg  = (int)(tid & 15);
  const long pix = tid >> 4;
  const int x = (int)(pix & (BGW - 1));
  const int y = (int)(pix >> 8);

  float vals[9];
  float fx[NINST], fy[NINST];
  int ix0[NINST], ix1[NINST], iy0[NINST], iy1[NINST];
  bool inb[NINST];
  bool covered = false;

  for (int i = 0; i < NINST; ++i) {
    const int l  = bbox[0 * NINST + i];
    const int t  = bbox[2 * NINST + i];
    const int rw = bbox[4 * NINST + i];
    const int rh = bbox[5 * NINST + i];
    const bool ib = (x >= l && x < l + rw && y >= t && y < t + rh);
    inb[i] = ib;
    covered = covered || ib;
    float v = 0.0f;
    if (ib) {
      const float syf = ((y - t) + 0.5f) * (64.0f / (float)rh) - 0.5f;
      const float sxf = ((x - l) + 0.5f) * (64.0f / (float)rw) - 0.5f;
      const int yf = (int)floorf(syf), xf = (int)floorf(sxf);
      const float wy = syf - (float)yf, wx = sxf - (float)xf;
      const int y0 = min(max(yf, 0), 63),     xq0 = min(max(xf, 0), 63);
      const int y1 = min(max(yf + 1, 0), 63), xq1 = min(max(xf + 1, 0), 63);
      iy0[i] = y0; iy1[i] = y1; ix0[i] = xq0; ix1[i] = xq1; fy[i] = wy; fx[i] = wx;
      const float* p = predI + (long)i * INH * INW;
      const float v00 = p[y0 * INW + xq0], v01 = p[y0 * INW + xq1];
      const float v10 = p[y1 * INW + xq0], v11 = p[y1 * INW + xq1];
      v = v00 * (1.f - wy) * (1.f - wx) + v01 * (1.f - wy) * wx +
          v10 * wy * (1.f - wx)         + v11 * wy * wx;
    }
    vals[i] = v;
  }
  vals[8] = predBG[pix] + (covered ? 0.0f : BIGV);

  float mx = vals[0];
  for (int i = 1; i < 9; ++i) mx = fmaxf(mx, vals[i]);
  float ex[9], ssum = 0.0f;
  for (int i = 0; i < 9; ++i) { ex[i] = __expf(vals[i] - mx); ssum += ex[i]; }
  const float inv = 1.0f / ssum;

  for (int cc = 0; cc < 8; ++cc) {
    const int c = cg * 8 + cc;
    float acc = ex[8] * inv * bgf[((long)c * BGH + y) * BGW + x];
    for (int i = 0; i < NINST; ++i) {
      if (!inb[i]) continue;
      const float* p = instf + ((long)i * CCH + c) * (INH * INW);
      const float wy = fy[i], wx = fx[i];
      const float v00 = p[iy0[i] * INW + ix0[i]], v01 = p[iy0[i] * INW + ix1[i]];
      const float v10 = p[iy1[i] * INW + ix0[i]], v11 = p[iy1[i] * INW + ix1[i]];
      const float s = v00 * (1.f - wy) * (1.f - wx) + v01 * (1.f - wy) * wx +
                      v10 * wy * (1.f - wx)         + v11 * wy * wx;
      acc += ex[i] * inv * s;
    }
    out[((long)c * BGH + y) * BGW + x] = acc;
  }
}

// ---------------------------------------------------------------------------
extern "C" void kernel_launch(void* const* d_in, const int* in_sizes, int n_in,
                              void* d_out, int out_size, void* d_ws, size_t ws_size,
                              hipStream_t stream) {
  const float* instf = (const float*)d_in[0];   // (8,128,64,64)
  const float* bgf   = (const float*)d_in[1];   // (1,128,256,256)
  const int*   bbox  = (const int*)  d_in[2];   // (6,8)
  // d_in[3] mrcnn_mask: unused by the reference
  const float* iW1 = (const float*)d_in[4];
  const float* ib1 = (const float*)d_in[5];
  const float* iW2 = (const float*)d_in[6];
  const float* ib2 = (const float*)d_in[7];
  const float* bW1 = (const float*)d_in[8];
  const float* bb1 = (const float*)d_in[9];
  const float* bW2 = (const float*)d_in[10];
  const float* bb2 = (const float*)d_in[11];
  float* out = (float*)d_out;

  char* ws = (char*)d_ws;
  size_t off = 0;
  auto take = [&](size_t bytes) {
    char* p = ws + off;
    off = (off + bytes + 255) & ~(size_t)255;
    return p;
  };
  const size_t bgAct = (size_t)BGH * BGW * CCH * sizeof(__bf16);   // 16 MB
  const size_t inAct = (size_t)NINST * INH * INW * CCH * sizeof(__bf16);
  __bf16* bg_hi = (__bf16*)take(bgAct);
  __bf16* bg_lo = (__bf16*)take(bgAct);
  __bf16* in_hi = (__bf16*)take(inAct);
  __bf16* in_lo = (__bf16*)take(inAct);
  __bf16* wI_hi = (__bf16*)take(PACKW * sizeof(__bf16));
  __bf16* wI_lo = (__bf16*)take(PACKW * sizeof(__bf16));
  __bf16* wB_hi = (__bf16*)take(PACKW * sizeof(__bf16));
  __bf16* wB_lo = (__bf16*)take(PACKW * sizeof(__bf16));
  float* bgc1  = (float*)take((size_t)BGH * BGW * UCH * sizeof(float));     // 33.5 MB
  float* inc1  = (float*)take((size_t)NINST * INH * INW * UCH * sizeof(float));
  float* predI = (float*)take((size_t)NINST * INH * INW * sizeof(float));
  float* predB = (float*)take((size_t)BGH * BGW * sizeof(float));
  (void)ws_size; (void)n_in; (void)in_sizes; (void)out_size;

  // K1: activation conversion (NCHW f32 -> HWC bf16 hi/lo)
  {
    long tot = (long)BGH * BGW * CCH;
    cvt_hwc_bf16<<<(unsigned)((tot + 255) / 256), 256, 0, stream>>>(
        bgf, bg_hi, bg_lo, 1, BGH, BGW);
    tot = (long)NINST * INH * INW * CCH;
    cvt_hwc_bf16<<<(unsigned)((tot + 255) / 256), 256, 0, stream>>>(
        instf, in_hi, in_lo, NINST, INH, INW);
  }
  // K2: weight packing
  pack_w<<<(PACKW + 255) / 256, 256, 0, stream>>>(iW1, wI_hi, wI_lo);
  pack_w<<<(PACKW + 255) / 256, 256, 0, stream>>>(bW1, wB_hi, wB_lo);

  // K3: WMMA conv1 + ReLU (TDM-staged weights)
  conv1_wmma<<<dim3(BGW / 64, BGH, 1), dim3(32, 4), 0, stream>>>(
      bg_hi, bg_lo, wB_hi, wB_lo, bb1, bgc1, BGH, BGW);
  conv1_wmma<<<dim3(INW / 64, INH, NINST), dim3(32, 4), 0, stream>>>(
      in_hi, in_lo, wI_hi, wI_lo, ib1, inc1, INH, INW);

  // K4: conv2 + ReLU -> mask planes
  conv2_relu<<<dim3((BGH * BGW) / 256, 1, 1), 256, 0, stream>>>(
      bgc1, bW2, bb2, predB, BGH, BGW);
  conv2_relu<<<dim3((INH * INW) / 256, 1, NINST), 256, 0, stream>>>(
      inc1, iW2, ib2, predI, INH, INW);

  // K5: fused resize/pad/softmax/blend -> output (1,128,256,256)
  {
    long tot = (long)BGH * BGW * 16;
    fuse_out<<<(unsigned)((tot + 255) / 256), 256, 0, stream>>>(
        instf, bgf, bbox, predI, predB, out);
  }
}
